// DotProductAttention_10136122818784
// MI455X (gfx1250) — matrix operand order repaired and checked by reading, not compiled, not run on hardware
//
#include <hip/hip_runtime.h>
#include <math.h>

// ---------------------------------------------------------------------------
// DotProductAttention for MI455X (gfx1250, wave32)
//   hidden_state:   [1, 64, 1024] f32   (d_in[0])
//   encoder_states: [64, 2048, 1024] f32 (d_in[1])
//   out:            [64, 1, 1024] f32
//
// Single fused pass over encoder_states (512 MB -> ~22us at 23.3 TB/s) with
// online softmax. Scores via v_wmma_f32_16x16x32_bf16 (bf16 mul, f32 acc);
// weighted sum in full f32 VALU (second touch of each 128KB chunk is L2-hot).
// This round: cheap bf16 packing (hw cvt_pk or v_perm) and wave0-only
// softmax bookkeeping with shfl reductions to slash VALU count.
// ---------------------------------------------------------------------------

typedef __attribute__((ext_vector_type(16))) __bf16 v16bf;
typedef __attribute__((ext_vector_type(8)))  float  v8f;

#define BATCH   64
#define S_TOTAL 2048
#define DDIM    1024
#define PARTS   8                     // blocks per batch
#define S_PART  (S_TOTAL / PARTS)     // 256 rows per block
#define CHUNK   32                    // rows per online-softmax chunk
#define NCHUNK  (S_PART / CHUNK)      // 8
#define NWAVES  8
#define BLOCK   256

#if __has_builtin(__builtin_amdgcn_cvt_pk_bf16_f32)
typedef __attribute__((ext_vector_type(2))) __bf16 v2bf;
__device__ __forceinline__ unsigned int pack2_bf16(float lo, float hi) {
  union { v2bf v; unsigned int u; } x;
  x.v = __builtin_amdgcn_cvt_pk_bf16_f32(lo, hi);
  return x.u;
}
#else
__device__ __forceinline__ unsigned int pack2_bf16(float lo, float hi) {
  // round-to-nearest (ties away from zero), then pack the two upper halves
  // with a single v_perm_b32.
  unsigned int a = __float_as_uint(lo) + 0x8000u;
  unsigned int b = __float_as_uint(hi) + 0x8000u;
  return __byte_perm(a, b, 0x7632u);
}
#endif

__device__ __forceinline__ unsigned short f2bf(float f) {
  unsigned int a = __float_as_uint(f);
  a = (a + 0x7FFFu + ((a >> 16) & 1u)) >> 16;   // RNE (h converted once)
  return (unsigned short)a;
}

union ABPack {
  v16bf v;
  unsigned int u[8];
  uint4 q[2];
};

__global__ __launch_bounds__(BLOCK) void attn_partial(
    const float* __restrict__ H, const float* __restrict__ E,
    float* __restrict__ wsM, float* __restrict__ wsL,
    float* __restrict__ wsAcc)
{
  __shared__ __align__(16) unsigned short lh[DDIM];  // h[b] as bf16
  __shared__ float psc[NWAVES][CHUNK];               // per-wave partial scores
  __shared__ float pw[CHUNK];                        // exp(score - m_new)
  __shared__ float s_scale;                          // acc rescale factor

  const int tid  = threadIdx.x;
  const int lane = tid & 31;
  const int wave = tid >> 5;
  const int b    = blockIdx.x / PARTS;
  const int part = blockIdx.x % PARTS;
  const int sbase = part * S_PART;

  // stage h[b] into LDS as bf16 (2 KB)
  for (int i = tid; i < DDIM; i += BLOCK) lh[i] = f2bf(H[b * DDIM + i]);
  __syncthreads();

  const int mrow = lane & 15;
  const int hi16 = lane >> 4;  // 0 for lanes 0-15, 1 for lanes 16-31
  const float* Eb = E + (size_t)b * S_TOTAL * DDIM;

  float  m_run = -INFINITY;   // meaningful only in wave 0 (uniform there)
  float  l_run = 0.0f;        // meaningful only in wave 0
  float4 acc   = make_float4(0.f, 0.f, 0.f, 0.f);
  const int d4 = tid * 4;     // this thread's 4 output dims

  for (int ch = 0; ch < NCHUNK; ++ch) {
    const int s0 = sbase + ch * CHUNK;

    // ---- score phase: this wave covers D-slice [wave*128, wave*128+128)
    v8f c0 = {};
    v8f c1 = {};
    const float* r0 = Eb + (size_t)(s0 + mrow) * DDIM;       // rows s0..s0+15
    const float* r1 = r0 + 16 * DDIM;                        // rows +16..+31
#pragma unroll
    for (int kk = 0; kk < 4; ++kk) {
      const int dbase = (wave * 4 + kk) * 32;
      const int dA    = dbase + (hi16 << 4);

      // A tiles: each lane takes 16 consecutive f32 of its row, cvt to bf16
      ABPack a0, a1, bb;
      const float4* p0 = (const float4*)(r0 + dA);
      const float4* p1 = (const float4*)(r1 + dA);
      float4 f0 = p0[0], f1 = p0[1], f2 = p0[2], f3 = p0[3];
      float4 g0 = p1[0], g1 = p1[1], g2 = p1[2], g3 = p1[3];
      a0.u[0] = pack2_bf16(f0.x, f0.y); a0.u[1] = pack2_bf16(f0.z, f0.w);
      a0.u[2] = pack2_bf16(f1.x, f1.y); a0.u[3] = pack2_bf16(f1.z, f1.w);
      a0.u[4] = pack2_bf16(f2.x, f2.y); a0.u[5] = pack2_bf16(f2.z, f2.w);
      a0.u[6] = pack2_bf16(f3.x, f3.y); a0.u[7] = pack2_bf16(f3.z, f3.w);
      a1.u[0] = pack2_bf16(g0.x, g0.y); a1.u[1] = pack2_bf16(g0.z, g0.w);
      a1.u[2] = pack2_bf16(g1.x, g1.y); a1.u[3] = pack2_bf16(g1.z, g1.w);
      a1.u[4] = pack2_bf16(g2.x, g2.y); a1.u[5] = pack2_bf16(g2.z, g2.w);
      a1.u[6] = pack2_bf16(g3.x, g3.y); a1.u[7] = pack2_bf16(g3.z, g3.w);

      // B = h broadcast into all 16 columns with matching K permutation
      const int off0 = dbase + (hi16 << 4);
      bb.q[0] = *(const uint4*)(&lh[off0]);
      bb.q[1] = *(const uint4*)(&lh[off0 + 8]);

      c0 = __builtin_amdgcn_wmma_f32_16x16x32_bf16(
          false, a0.v, false, bb.v, (short)0, c0, false, false);
      c1 = __builtin_amdgcn_wmma_f32_16x16x32_bf16(
          false, a1.v, false, bb.v, (short)0, c1, false, false);
    }

    // C layout: lanes 0-15 hold M=0..7 in regs 0..7 (col = lane), lanes 16-31
    // hold M=8..15. B broadcast => all columns equal; take lane 0 / lane 16.
    if (lane == 0) {
#pragma unroll
      for (int r = 0; r < 8; ++r) {
        psc[wave][r]      = c0[r];
        psc[wave][16 + r] = c1[r];
      }
    } else if (lane == 16) {
#pragma unroll
      for (int r = 0; r < 8; ++r) {
        psc[wave][8 + r]  = c0[r];
        psc[wave][24 + r] = c1[r];
      }
    }
    __syncthreads();  // B1: psc complete

    // ---- wave 0: reduce partial scores + online softmax update
    if (wave == 0) {
      float sv = 0.f;
#pragma unroll
      for (int w = 0; w < NWAVES; ++w) sv += psc[w][lane];
      // wave-wide max of the 32 chunk scores
      float cm = sv;
#pragma unroll
      for (int off = 16; off > 0; off >>= 1)
        cm = fmaxf(cm, __shfl_xor(cm, off, 32));
      const float m_new = fmaxf(m_run, cm);
      const float p = __expf(sv - m_new);
      pw[lane] = p;
      float cs = p;
#pragma unroll
      for (int off = 16; off > 0; off >>= 1)
        cs += __shfl_xor(cs, off, 32);
      const float scale = __expf(m_run - m_new);  // 0 on first chunk (-inf)
      l_run = l_run * scale + cs;
      m_run = m_new;
      if (lane == 0) s_scale = scale;
    }
    __syncthreads();  // B2: pw + s_scale ready

    // ---- weighted accumulation over this chunk (rows are L2/WGP$-hot)
    const float scale = s_scale;
    acc.x *= scale; acc.y *= scale; acc.z *= scale; acc.w *= scale;
    const float* Er = Eb + (size_t)s0 * DDIM + d4;
#pragma unroll 4
    for (int s = 0; s < CHUNK; ++s) {
      const float w  = pw[s];
      const float4 e = *(const float4*)(Er + (size_t)s * DDIM);
      acc.x = fmaf(w, e.x, acc.x);
      acc.y = fmaf(w, e.y, acc.y);
      acc.z = fmaf(w, e.z, acc.z);
      acc.w = fmaf(w, e.w, acc.w);
    }
    __syncthreads();  // B3: protect psc/pw/s_scale for next chunk
  }

  const int idx = b * PARTS + part;
  *(float4*)(wsAcc + (size_t)idx * DDIM + d4) = acc;
  if (tid == 0) { wsM[idx] = m_run; wsL[idx] = l_run; }
}

// Combine the 8 partials per batch:
//   out = sum_p e^{m_p-M} acc_p / sum_p e^{m_p-M} l_p
__global__ __launch_bounds__(BLOCK) void attn_combine(
    const float* __restrict__ wsM, const float* __restrict__ wsL,
    const float* __restrict__ wsAcc, float* __restrict__ out)
{
  const int b   = blockIdx.x;
  const int tid = threadIdx.x;

  float M = -INFINITY;
#pragma unroll
  for (int p = 0; p < PARTS; ++p) M = fmaxf(M, wsM[b * PARTS + p]);
  float L = 0.f;
#pragma unroll
  for (int p = 0; p < PARTS; ++p)
    L += wsL[b * PARTS + p] * __expf(wsM[b * PARTS + p] - M);

  const int d4 = tid * 4;
  float ax = 0.f, ay = 0.f, az = 0.f, aw = 0.f;
#pragma unroll
  for (int p = 0; p < PARTS; ++p) {
    const float sp = __expf(wsM[b * PARTS + p] - M);
    const float4 a =
        *(const float4*)(wsAcc + (size_t)(b * PARTS + p) * DDIM + d4);
    ax = fmaf(sp, a.x, ax);
    ay = fmaf(sp, a.y, ay);
    az = fmaf(sp, a.z, az);
    aw = fmaf(sp, a.w, aw);
  }
  const float inv = 1.0f / L;
  *(float4*)(out + (size_t)b * DDIM + d4) =
      make_float4(ax * inv, ay * inv, az * inv, aw * inv);
}

extern "C" void kernel_launch(void* const* d_in, const int* in_sizes, int n_in,
                              void* d_out, int out_size, void* d_ws,
                              size_t ws_size, hipStream_t stream) {
  (void)in_sizes; (void)n_in; (void)out_size; (void)ws_size;
  const float* H = (const float*)d_in[0];   // [1,64,1024]
  const float* E = (const float*)d_in[1];   // [64,2048,1024]
  float* out = (float*)d_out;               // [64,1,1024]

  // workspace layout (floats): M[512] | L[512] | Acc[512*1024]  (~2.1 MB)
  float* ws    = (float*)d_ws;
  const int nIdx = BATCH * PARTS;           // 512
  float* wsM   = ws;
  float* wsL   = ws + nIdx;
  float* wsAcc = ws + 2 * nIdx;

  attn_partial<<<dim3(BATCH * PARTS), dim3(BLOCK), 0, stream>>>(
      H, E, wsM, wsL, wsAcc);
  attn_combine<<<dim3(BATCH), dim3(BLOCK), 0, stream>>>(wsM, wsL, wsAcc, out);
}
